// CausalSelfAttention_51874615001434
// MI455X (gfx1250) — matrix operand order repaired and checked by reading, not compiled, not run on hardware
//
#include <hip/hip_runtime.h>

// ---------------------------------------------------------------------------
// Causal self-attention with RoPE for MI455X (gfx1250, wave32, WMMA).
// Pipeline: cvt->bf16, QKV GEMM (WMMA bf16), RoPE split (V transposed),
// flash-attention (WMMA bf16, 64-key tiles, online softmax f32),
// output GEMM (f32 out).
// ---------------------------------------------------------------------------

typedef __attribute__((ext_vector_type(16))) __bf16          v16bf;
typedef __attribute__((ext_vector_type(8)))  float           v8f;
typedef __attribute__((ext_vector_type(16))) unsigned short  v16u;
typedef __attribute__((ext_vector_type(8)))  unsigned short  v8u;

#define D_MODEL 1024
#define SEQ     4096
#define NH      16
#define DK      64
#define PSTR    72   // P-tile LDS row stride (halves): 144B, conflict-free

__device__ __forceinline__ unsigned short f32_bf16(float f) {
  union { float f; unsigned u; } v; v.f = f;
  unsigned u = v.u;
  u += 0x7fffu + ((u >> 16) & 1u);           // round-to-nearest-even
  return (unsigned short)(u >> 16);
}
__device__ __forceinline__ float bf16_f32(unsigned short h) {
  union { unsigned u; float f; } v; v.u = ((unsigned)h) << 16;
  return v.f;
}

// Two 16B loads -> one 32B fragment register pair, no element moves.
struct alignas(32) v8u_pair { v8u lo; v8u hi; };
__device__ __forceinline__ v16u ldfrag(const unsigned short* p, int hi_off_halves) {
  v8u_pair pr;
  pr.lo = *reinterpret_cast<const v8u*>(p);
  pr.hi = *reinterpret_cast<const v8u*>(p + hi_off_halves);
  return __builtin_bit_cast(v16u, pr);
}

__device__ __forceinline__ v8f wmma_bf16(v16u a, v16u b, v8f c) {
  return __builtin_amdgcn_wmma_f32_16x16x32_bf16(
      false, __builtin_bit_cast(v16bf, a),
      false, __builtin_bit_cast(v16bf, b),
      (short)0, c, false, false);
}

// ---------------------------------------------------------------------------
__global__ __launch_bounds__(256) void cvt_f32_bf16_k(
    const float* __restrict__ src, unsigned short* __restrict__ dst, int n) {
  int i = blockIdx.x * 256 + threadIdx.x;
  if (i < n) dst[i] = f32_bf16(src[i]);
}

// ---------------------------------------------------------------------------
// C[M,N] = A[M,K] * B[N,K]^T   (A,B bf16 row-major; C bf16 or f32)
// Block tile 128x128, 8 waves (4x2), wave tile 32x64, K step 32.
// LDS row stride 40 halves (80 B) -> bank-conflict-free 16B fragment loads.
template <bool OUT_BF16>
__global__ __launch_bounds__(256) void gemm_xwT(
    const unsigned short* __restrict__ A,
    const unsigned short* __restrict__ B,
    void* __restrict__ C, int M, int N, int K) {
  __shared__ alignas(16) unsigned short As[128 * 40];
  __shared__ alignas(16) unsigned short Bs[128 * 40];
  const int tid  = threadIdx.x;
  const int lane = tid & 31;
  const int wave = tid >> 5;
  const int wm   = wave & 3;           // 4 M-slices of 32 rows
  const int wn   = wave >> 2;          // 2 N-slices of 64 cols
  const int m0   = blockIdx.y * 128;
  const int n0   = blockIdx.x * 128;
  const int arow = lane & 15;          // A-frag row / B-frag col
  const int kb   = (lane >> 4) * 8;    // A-frag K base within 32
  const int kbb  = (lane >> 4) * 16;   // unused; B frag uses kb

  v8f acc[2][4] = {};

  for (int kk = 0; kk < K; kk += 32) {
    // Cooperative stage: 128x32 halves each for A and B, 16B vector loads.
#pragma unroll
    for (int j = 0; j < 2; ++j) {
      int linear = tid + j * 256;           // 0..511
      int row = linear >> 2;                // 0..127
      int seg = (linear & 3) * 8;           // 0,8,16,24 halves
      *reinterpret_cast<uint4*>(&As[row * 40 + seg]) =
          *reinterpret_cast<const uint4*>(&A[(size_t)(m0 + row) * K + kk + seg]);
      *reinterpret_cast<uint4*>(&Bs[row * 40 + seg]) =
          *reinterpret_cast<const uint4*>(&B[(size_t)(n0 + row) * K + kk + seg]);
    }
    if (kk + 32 < K) {                      // prefetch next K tile
      __builtin_prefetch(&A[(size_t)(m0 + (tid >> 1)) * K + kk + 32], 0, 0);
      __builtin_prefetch(&B[(size_t)(n0 + (tid >> 1)) * K + kk + 32], 0, 0);
    }
    __syncthreads();

    v16u af[2], bf[4];
#pragma unroll
    for (int i = 0; i < 2; ++i)             // A frag: K=kb..kb+7 , kb+16..kb+23
      af[i] = ldfrag(&As[(wm * 32 + i * 16 + arow) * 40 + kb], 16);
#pragma unroll
    for (int j = 0; j < 4; ++j)             // B frag: K=kb..kb+7 , kb+16..kb+23
      bf[j] = ldfrag(&Bs[(wn * 64 + j * 16 + arow) * 40 + kb], 16);
#pragma unroll
    for (int i = 0; i < 2; ++i)
#pragma unroll
      for (int j = 0; j < 4; ++j)
        acc[i][j] = wmma_bf16(af[i], bf[j], acc[i][j]);
    __syncthreads();
  }

  const int crow = (lane >> 4) * 8;         // C layout: M = r + crow, N = ccol
  const int ccol = lane & 15;
#pragma unroll
  for (int i = 0; i < 2; ++i)
#pragma unroll
    for (int j = 0; j < 4; ++j)
#pragma unroll
      for (int r = 0; r < 8; ++r) {
        int gr = m0 + wm * 32 + i * 16 + crow + r;
        int gc = n0 + wn * 64 + j * 16 + ccol;
        if (OUT_BF16)
          ((unsigned short*)C)[(size_t)gr * N + gc] = f32_bf16(acc[i][j][r]);
        else
          ((float*)C)[(size_t)gr * N + gc] = acc[i][j][r];
      }
}

// ---------------------------------------------------------------------------
// RoPE on Q,K + split V. qkv: [S, 3*D] bf16.
// Qr/Kr: [S, D] bf16 (col = h*64 + d).  Vt: TRANSPOSED [D, S] bf16 so the
// flash-attention PV B-fragment is contiguous along the key dimension.
__global__ __launch_bounds__(256) void rope_split(
    const unsigned short* __restrict__ qkv,
    unsigned short* __restrict__ Qr,
    unsigned short* __restrict__ Kr,
    unsigned short* __restrict__ Vt) {
  const int idx = blockIdx.x * 256 + threadIdx.x;   // s*D + c
  const int s = idx >> 10;
  const int c = idx & (D_MODEL - 1);
  const int d = c & (DK - 1);
  const size_t base = (size_t)s * (3 * D_MODEL);
  const float q = bf16_f32(qkv[base + c]);
  const float k = bf16_f32(qkv[base + D_MODEL + c]);
  const int partner = (d < 32) ? (c + 32) : (c - 32);
  float qr = bf16_f32(qkv[base + partner]);
  float kr = bf16_f32(qkv[base + D_MODEL + partner]);
  if (d < 32) { qr = -qr; kr = -kr; }               // rotate_half
  const int i = d & 31;
  const float invf = expf(-(float)i * (9.210340371976184f / 32.0f)); // 1e4^(-2i/64)
  const float ang = (float)s * invf;
  const float cs = cosf(ang), sn = sinf(ang);
  Qr[idx] = f32_bf16(q * cs + qr * sn);
  Kr[idx] = f32_bf16(k * cs + kr * sn);
  Vt[(size_t)c * SEQ + s] = qkv[base + 2 * D_MODEL + c];
}

// ---------------------------------------------------------------------------
// Flash attention, causal. One wave per 16-query tile; key tiles of 64 to
// amortize softmax reductions/rescaling over 2x the WMMA work.
// grid = (SEQ/64, NH), block = 128 (4 independent waves).
__global__ __launch_bounds__(128) void attn_fa(
    const unsigned short* __restrict__ Qr,
    const unsigned short* __restrict__ Kr,
    const unsigned short* __restrict__ Vt,   // [D, S] transposed
    unsigned short* __restrict__ AO) {
  __shared__ alignas(16) unsigned short Pl[4][16 * PSTR];  // per-wave 16x64 P
  const int lane  = threadIdx.x & 31;
  const int wave  = threadIdx.x >> 5;
  const int h     = blockIdx.y;
  const int qbase = blockIdx.x * 64 + wave * 16;
  const size_t hoff = (size_t)h * DK;
  const int arow = lane & 15;
  const int kb   = (lane >> 4) * 8;
  const int kbb  = (lane >> 4) * 16;
  const int ccol = lane & 15;
  const int crow = (lane >> 4) * 8;
  unsigned short* Pw = &Pl[wave][0];

  // Q fragments: 16x64 = two 16x32 A-frags, loaded once.
  v16u qf[2];
#pragma unroll
  for (int c = 0; c < 2; ++c)
    qf[c] = ldfrag(&Qr[(size_t)(qbase + arow) * D_MODEL + hoff + c * 32 + kb], 16);

  // Per-dim base pointers for the transposed-V fragments (loop-invariant).
  const unsigned short* vbase[4];
#pragma unroll
  for (int j = 0; j < 4; ++j)
    vbase[j] = &Vt[(hoff + j * 16 + ccol) * (size_t)SEQ + kb];

  v8f O[4] = {};
  float m[8], l[8];
#pragma unroll
  for (int r = 0; r < 8; ++r) { m[r] = -1e30f; l[r] = 0.f; }

  const int nkt = (qbase + 16 + 63) >> 6;   // 64-key tiles covering causal span
  for (int kt = 0; kt < nkt; ++kt) {
    const int k0 = kt * 64;
    // ---- scores: S[t] = Q(16x64) * K^T(64x16), key groups t = 0..3
    v8f S[4] = {};
#pragma unroll
    for (int t = 0; t < 4; ++t) {
      const int key = k0 + t * 16 + ccol;   // B-frag column -> key row
      const unsigned short* kp = &Kr[(size_t)key * D_MODEL + hoff + kb];
#pragma unroll
      for (int c = 0; c < 2; ++c)
        S[t] = wmma_bf16(qf[c], ldfrag(kp + c * 32, 16), S[t]);
    }
    // ---- online softmax (C layout: row = crow + r, col = t*16 + ccol)
    float p[4][8];
#pragma unroll
    for (int r = 0; r < 8; ++r) {
      const int q = qbase + crow + r;
      float v[4];
#pragma unroll
      for (int t = 0; t < 4; ++t) {
        v[t] = S[t][r] * 0.125f;                    // 1/sqrt(64)
        if (k0 + t * 16 + ccol > q) v[t] = -1e9f;
      }
      float mx = fmaxf(fmaxf(v[0], v[1]), fmaxf(v[2], v[3]));
#pragma unroll
      for (int d = 1; d < 16; d <<= 1) mx = fmaxf(mx, __shfl_xor(mx, d, 32));
      const float mnew = fmaxf(m[r], mx);
      const float sc = expf(m[r] - mnew);
      float rs = 0.f;
#pragma unroll
      for (int t = 0; t < 4; ++t) { p[t][r] = expf(v[t] - mnew); rs += p[t][r]; }
#pragma unroll
      for (int d = 1; d < 16; d <<= 1) rs += __shfl_xor(rs, d, 32);
      l[r] = l[r] * sc + rs;
      m[r] = mnew;
#pragma unroll
      for (int j = 0; j < 4; ++j) O[j][r] *= sc;
    }
    // ---- re-layout P (C layout -> A layout) via per-wave LDS tile
#pragma unroll
    for (int r = 0; r < 8; ++r)
#pragma unroll
      for (int t = 0; t < 4; ++t)
        Pw[(crow + r) * PSTR + t * 16 + ccol] = f32_bf16(p[t][r]);
    v16u pf[2];
#pragma unroll
    for (int c = 0; c < 2; ++c)
      pf[c] = ldfrag(Pw + arow * PSTR + c * 32 + kb, 16);
    // ---- O(16x64) += P(16x64) * V(64x64); V frag contiguous via Vt
#pragma unroll
    for (int j = 0; j < 4; ++j)
#pragma unroll
      for (int c = 0; c < 2; ++c)
        O[j] = wmma_bf16(pf[c], ldfrag(vbase[j] + k0 + c * 32, 16), O[j]);
  }
  // ---- normalize and store (bf16, [S, H*dk] head-major columns)
#pragma unroll
  for (int r = 0; r < 8; ++r) {
    const float inv = 1.0f / l[r];
    const size_t rowoff = (size_t)(qbase + crow + r) * D_MODEL + hoff;
#pragma unroll
    for (int j = 0; j < 4; ++j)
      AO[rowoff + j * 16 + ccol] = f32_bf16(O[j][r] * inv);
  }
}

// ---------------------------------------------------------------------------
extern "C" void kernel_launch(void* const* d_in, const int* in_sizes, int n_in,
                              void* d_out, int out_size, void* d_ws, size_t ws_size,
                              hipStream_t stream) {
  const float* x    = (const float*)d_in[0];   // [1,4096,1024]
  const float* Wqkv = (const float*)d_in[1];   // [3072,1024]
  const float* Wo   = (const float*)d_in[2];   // [1024,1024]
  float* out = (float*)d_out;                  // [1,4096,1024]

  unsigned short* Xb    = (unsigned short*)d_ws;                     // 4096x1024
  unsigned short* Wqkvb = Xb    + (size_t)SEQ * D_MODEL;             // 3072x1024
  unsigned short* Wob   = Wqkvb + (size_t)3 * D_MODEL * D_MODEL;     // 1024x1024
  unsigned short* QKVb  = Wob   + (size_t)D_MODEL * D_MODEL;         // 4096x3072
  unsigned short* Qr    = QKVb  + (size_t)SEQ * 3 * D_MODEL;         // 4096x1024
  unsigned short* Kr    = Qr    + (size_t)SEQ * D_MODEL;
  unsigned short* Vt    = Kr    + (size_t)SEQ * D_MODEL;             // [D, S]
  unsigned short* AOb   = Vt    + (size_t)SEQ * D_MODEL;

  cvt_f32_bf16_k<<<(SEQ * D_MODEL) / 256, 256, 0, stream>>>(x, Xb, SEQ * D_MODEL);
  cvt_f32_bf16_k<<<(3 * D_MODEL * D_MODEL) / 256, 256, 0, stream>>>(
      Wqkv, Wqkvb, 3 * D_MODEL * D_MODEL);
  cvt_f32_bf16_k<<<(D_MODEL * D_MODEL) / 256, 256, 0, stream>>>(
      Wo, Wob, D_MODEL * D_MODEL);

  gemm_xwT<true><<<dim3(3 * D_MODEL / 128, SEQ / 128), 256, 0, stream>>>(
      Xb, Wqkvb, (void*)QKVb, SEQ, 3 * D_MODEL, D_MODEL);

  rope_split<<<(SEQ * D_MODEL) / 256, 256, 0, stream>>>(QKVb, Qr, Kr, Vt);

  attn_fa<<<dim3(SEQ / 64, NH), 128, 0, stream>>>(Qr, Kr, Vt, AOb);

  gemm_xwT<false><<<dim3(D_MODEL / 128, SEQ / 128), 256, 0, stream>>>(
      AOb, Wob, (void*)out, SEQ, D_MODEL, D_MODEL);
}
